// AttDecode_2250562863941
// MI455X (gfx1250) — matrix operand order repaired and checked by reading, not compiled, not run on hardware
//
#include <hip/hip_runtime.h>
#include <hip/hip_bf16.h>

// ---------------------------------------------------------------------------
// AttDecode forward for MI455X (gfx1250, wave32, WMMA).
//
// Key optimization: in the reference, `h` is never updated across layers and
// `out` is overwritten each iteration, so the result depends ONLY on layer
// ST-1 = 5. Layers 0..4 are dead code and are skipped (6x work reduction).
//
// All GEMMs run on v_wmma_f32_16x16x32_bf16 (bf16 in, f32 accumulate).
// Weights are transposed + zero-padded to bf16 once per launch so every
// B-fragment lane load is a contiguous, UNGUARDED 16B chunk (all BT buffers
// padded to full 32-wide tile coverage -> branch-free inner loop).
// ---------------------------------------------------------------------------

typedef __attribute__((ext_vector_type(16))) __bf16 v16bf;
typedef __attribute__((ext_vector_type(8)))  float  v8f;

#define B_  32
#define L_  512
#define E_  200
#define D_  200
#define H_  8
#define KP  224   // pad for E/D dims (multiple of 32) - used for K and BT rows
#define HD  1600  // H*D
#define BL  16384 // B*L
#define BH  256   // B*H

static __device__ inline unsigned short f32_to_bf16(float f) {
  unsigned int u = __builtin_bit_cast(unsigned int, f);
  unsigned int r = (u + 0x7FFFu + ((u >> 16) & 1u)) >> 16;
  return (unsigned short)r;
}

static __device__ inline v8f wmma_bf16(v16bf a, v16bf b, v8f c) {
  return __builtin_amdgcn_wmma_f32_16x16x32_bf16(
      /*neg_a=*/false, a, /*neg_b=*/false, b,
      /*c_mod=*/(short)0, c, /*reuse_a=*/false, /*reuse_b=*/false);
}

static __device__ inline v16bf make_v16(uint4 lo, uint4 hi) {
  union { uint4 u[2]; v16bf v; } t;
  t.u[0] = lo; t.u[1] = hi;
  return t.v;
}

// A fragment, 16x32 bf16 per ISA layout:
//   lane m = lane&15; lanes 0-15 hold K 0..7 & 16..23, lanes 16-31 K 8..15 & 24..31
static __device__ inline v16bf loadA(const unsigned short* A, long lda,
                                     int m0, int k0, int lane) {
  int ml = lane & 15, g = lane >> 4;
  const unsigned short* row = A + (long)(m0 + ml) * lda + k0;
  uint4 lo = *(const uint4*)(row + g * 8);
  uint4 hi = *(const uint4*)(row + 16 + g * 8);
  return make_v16(lo, hi);
}

// B fragment, 32x16 bf16. B is stored pre-transposed (BT[n][k]) so each lane
// reads K values of its column n: lanes 0-15 K 0..7 & 16..23, 16-31 K 8..15 & 24..31.
// Unguarded: all BT buffers are allocated/padded to full tile coverage.
static __device__ inline v16bf loadB(const unsigned short* BT, long ldb,
                                     int n0, int k0, int lane) {
  int nl = lane & 15, g = lane >> 4;
  const unsigned short* p = BT + (long)(n0 + nl) * ldb + k0 + g * 8;
  uint4 lo = *(const uint4*)p;
  uint4 hi = *(const uint4*)(p + 16);
  return make_v16(lo, hi);
}

struct GemmArgs {
  const unsigned short* A;  long lda, zsA;
  const unsigned short* BT; long ldb, zsB;
  int M, Ntiles, Kpad, Nstore;
  const float* bias; long biasZs; int biasN;
  float* outF; unsigned short* outB; long ldo, zsO;
  const float* res; long ldr;
  int H, L;
};

// Epilogues:
// 0: q/k  -> bf16 [b,h,l,224] (+bias)
// 1: v    -> bf16 transposed [b,h,d(224),l] (+bias; rows >=200 exact zeros)
// 2: S    -> f32 [z,m,n]
// 3: attn -> bf16 c[b*l, h*200+d]
// 4: Wf   -> f32 zpre = acc + bias + residual(h)
// 5: FFN1 -> bf16 relu(acc+bias), full 224 cols (pads compute to 0)
// 6: FFN2 -> f32 f2 = acc + bias + residual(z)
template <int EPI>
static __device__ inline void store16(const GemmArgs& P, int z, int m0, int n0,
                                      int lane, v8f c) {
  int nl = lane & 15, g = lane >> 4;
  int n = n0 + nl;
  const float* biasp = P.bias ? P.bias + (long)z * P.biasZs : nullptr;
  float bi = (biasp && n < P.biasN) ? biasp[n] : 0.f;
#pragma unroll
  for (int r = 0; r < 8; ++r) {
    int m = m0 + g * 8 + r;
    float v = c[r] + bi;
    if (EPI == 0) {
      int b = m / P.L, l = m % P.L;
      P.outB[(((long)b * P.H + z) * P.L + l) * P.ldo + n] = f32_to_bf16(v);
    } else if (EPI == 1) {
      int b = m / P.L, l = m % P.L;
      P.outB[(((long)b * P.H + z) * P.Nstore + n) * P.ldo + l] = f32_to_bf16(v);
    } else if (EPI == 2) {
      P.outF[(long)z * P.zsO + (long)m * P.ldo + n] = v;
    } else if (EPI == 3) {
      if (n < P.Nstore) {
        int b = z / P.H, h = z % P.H;
        P.outB[((long)b * P.L + m) * P.ldo + (long)h * P.Nstore + n] = f32_to_bf16(v);
      }
    } else if (EPI == 4 || EPI == 6) {
      if (n < P.Nstore)
        P.outF[(long)m * P.ldo + n] = v + P.res[(long)m * P.ldr + n];
    } else if (EPI == 5) {
      P.outB[(long)m * P.ldo + n] = f32_to_bf16(v > 0.f ? v : 0.f);
    }
  }
}

// 256 threads = 8 wave32; each wave computes one 32x32 output tile (4 WMMAs
// per 32-wide K step, A/B fragments each reused twice). Branch-free K loop.
template <int EPI>
__global__ __launch_bounds__(256) void gemm_wmma_k(GemmArgs P) {
  int wid = threadIdx.x >> 5, lane = threadIdx.x & 31;
  int z = blockIdx.z;
  int Mt = P.M >> 5;
  int tid = blockIdx.x * 8 + wid;
  if (tid >= Mt * P.Ntiles) return;
  int tm = tid % Mt, tn = tid / Mt;
  int m0 = tm << 5, n0 = tn << 5;
  const unsigned short* A  = P.A  + (long)z * P.zsA;
  const unsigned short* BT = P.BT + (long)z * P.zsB;
  v8f c00 = {}, c01 = {}, c10 = {}, c11 = {};
  for (int k = 0; k < P.Kpad; k += 32) {
    v16bf a0 = loadA(A, P.lda, m0, k, lane);
    v16bf a1 = loadA(A, P.lda, m0 + 16, k, lane);
    v16bf b0 = loadB(BT, P.ldb, n0, k, lane);
    v16bf b1 = loadB(BT, P.ldb, n0 + 16, k, lane);
    c00 = wmma_bf16(a0, b0, c00);
    c01 = wmma_bf16(a0, b1, c01);
    c10 = wmma_bf16(a1, b0, c10);
    c11 = wmma_bf16(a1, b1, c11);
  }
  store16<EPI>(P, z, m0,      n0,      lane, c00);
  store16<EPI>(P, z, m0,      n0 + 16, lane, c01);
  store16<EPI>(P, z, m0 + 16, n0,      lane, c10);
  store16<EPI>(P, z, m0 + 16, n0 + 16, lane, c11);
}

// ---------------- embedding + positional encoding -------------------------
__global__ __launch_bounds__(256) void embed_k(const int* __restrict__ y,
                                               const float* __restrict__ emb,
                                               float* __restrict__ h,
                                               unsigned short* __restrict__ hbf) {
  int b = blockIdx.x, t = threadIdx.x;
  __shared__ int cnt[256];
  int c = (y[b * L_ + t] == 0) + (y[b * L_ + 256 + t] == 0);
  cnt[t] = c;
  __syncthreads();
  for (int o = 128; o > 0; o >>= 1) {
    if (t < o) cnt[t] += cnt[t + o];
    __syncthreads();
  }
  int off = cnt[0] & 255;  // .byte() wrap
  for (int i = t; i < L_ * KP; i += 256) {
    int l = i / KP, e = i % KP;
    long row = (long)b * L_ + l;
    if (e >= E_) { hbf[row * KP + e] = 0; continue; }
    int yv = y[b * L_ + l];
    float v = emb[(long)yv * E_ + e];
    float ii = (float)(l - off);
    if (ii >= 0.f) {
      float ang = ii / powf(1000.f, ii / (float)E_);
      v += (e & 1) ? sinf(ang) : cosf(ang);
    }
    h[row * E_ + e] = v;
    hbf[row * KP + e] = f32_to_bf16(v);
  }
}

// ---------------- softmax over rows of S, emits bf16 P ---------------------
__global__ __launch_bounds__(256) void softmax_k(const float* __restrict__ S,
                                                 unsigned short* __restrict__ P) {
  int z = blockIdx.y, row = blockIdx.x, t = threadIdx.x;
  const float* s = S + ((long)z * L_ + row) * L_;
  unsigned short* p = P + ((long)z * L_ + row) * L_;
  const float inv = 0.07071067811865475f;  // 1/sqrt(200)
  __shared__ float red[256];
  float a = s[t] * inv, b = s[t + 256] * inv;
  red[t] = fmaxf(a, b);
  __syncthreads();
  for (int o = 128; o > 0; o >>= 1) {
    if (t < o) red[t] = fmaxf(red[t], red[t + o]);
    __syncthreads();
  }
  float m = red[0];
  __syncthreads();
  float e0 = __expf(a - m), e1 = __expf(b - m);
  red[t] = e0 + e1;
  __syncthreads();
  for (int o = 128; o > 0; o >>= 1) {
    if (t < o) red[t] += red[t + o];
    __syncthreads();
  }
  float rs = 1.f / red[0];
  p[t] = f32_to_bf16(e0 * rs);
  p[t + 256] = f32_to_bf16(e1 * rs);
}

// ---------------- joint (L,feat) layer norm per batch ----------------------
__global__ __launch_bounds__(256) void ln_k(const float* __restrict__ in, int ldin,
                                            float* __restrict__ outF, int ldoF,
                                            unsigned short* __restrict__ outB, int ldoB,
                                            int rows, int N, int Npad) {
  int b = blockIdx.x, t = threadIdx.x;
  const float* base = in + (long)b * rows * ldin;
  long total = (long)rows * N;
  float s = 0.f, ss = 0.f;
  for (long i = t; i < total; i += 256) {
    int r = (int)(i / N), c = (int)(i % N);
    float v = base[(long)r * ldin + c];
    s += v; ss += v * v;
  }
  __shared__ float rs_[256], rss_[256];
  rs_[t] = s; rss_[t] = ss;
  __syncthreads();
  for (int o = 128; o > 0; o >>= 1) {
    if (t < o) { rs_[t] += rs_[t + o]; rss_[t] += rss_[t + o]; }
    __syncthreads();
  }
  float mean = rs_[0] / (float)total;
  float var = rss_[0] / (float)total - mean * mean;
  float rstd = rsqrtf(var + 1e-5f);
  for (long i = t; i < total; i += 256) {
    int r = (int)(i / N), c = (int)(i % N);
    float v = (base[(long)r * ldin + c] - mean) * rstd;
    if (outF) outF[((long)b * rows + r) * ldoF + c] = v;
    if (outB) outB[((long)b * rows + r) * ldoB + c] = f32_to_bf16(v);
  }
  if (outB && Npad > N) {
    long padTot = (long)rows * (Npad - N);
    for (long i = t; i < padTot; i += 256) {
      int r = (int)(i / (Npad - N)), c = N + (int)(i % (Npad - N));
      outB[((long)b * rows + r) * ldoB + c] = 0;
    }
  }
}

// ---------------- weight transpose/pad/convert to bf16 ---------------------
__global__ __launch_bounds__(256) void conv_qkv_k(const float* __restrict__ W,
                                                  unsigned short* __restrict__ WT) {
  // W: [H][E][D] slice; WT: [H][KP(d)][KP(e)], WT[h][d][e] = W[h][e][d]
  int idx = blockIdx.x * 256 + threadIdx.x;
  if (idx >= H_ * KP * KP) return;
  int h = idx / (KP * KP), rem = idx % (KP * KP);
  int d = rem / KP, e = rem % KP;
  float v = (d < D_ && e < E_) ? W[((long)h * E_ + e) * D_ + d] : 0.f;
  WT[idx] = f32_to_bf16(v);
}

__global__ __launch_bounds__(256) void conv_wf_k(const float* __restrict__ W,
                                                 unsigned short* __restrict__ WT) {
  // W: [HD][D]; WT: [KP][HD], WT[n][k] = W[k][n] (rows >= 200 zero)
  int idx = blockIdx.x * 256 + threadIdx.x;
  if (idx >= KP * HD) return;
  int n = idx / HD, k = idx % HD;
  float v = (n < D_) ? W[(long)k * D_ + n] : 0.f;
  WT[idx] = f32_to_bf16(v);
}

__global__ __launch_bounds__(256) void conv_sq_k(const float* __restrict__ W,
                                                 unsigned short* __restrict__ WT) {
  // W: [D][D]; WT: [KP][KP], WT[n][k] = W[k][n]
  int idx = blockIdx.x * 256 + threadIdx.x;
  if (idx >= KP * KP) return;
  int n = idx / KP, k = idx % KP;
  float v = (n < D_ && k < D_) ? W[(long)k * D_ + n] : 0.f;
  WT[idx] = f32_to_bf16(v);
}

// ---------------------------------------------------------------------------
extern "C" void kernel_launch(void* const* d_in, const int* in_sizes, int n_in,
                              void* d_out, int out_size, void* d_ws, size_t ws_size,
                              hipStream_t stream) {
  (void)in_sizes; (void)n_in; (void)out_size; (void)ws_size;
  const int*   y   = (const int*)d_in[0];
  const float* emb = (const float*)d_in[2];
  const float* Wq  = (const float*)d_in[3];
  const float* bq  = (const float*)d_in[4];
  const float* Wk  = (const float*)d_in[5];
  const float* bk  = (const float*)d_in[6];
  const float* Wv  = (const float*)d_in[7];
  const float* bv  = (const float*)d_in[8];
  const float* Wf  = (const float*)d_in[9];
  const float* bff = (const float*)d_in[10];
  const float* W1  = (const float*)d_in[11];
  const float* b1  = (const float*)d_in[12];
  const float* W2  = (const float*)d_in[13];
  const float* b2  = (const float*)d_in[14];

  char* ws = (char*)d_ws;
  size_t off = 0;
  auto take = [&](size_t bytes) -> char* {
    size_t a = (off + 255) & ~(size_t)255;
    off = a + bytes;
    return ws + a;
  };
  float*          h    = (float*)take((size_t)BL * E_ * 4);
  unsigned short* hbf  = (unsigned short*)take((size_t)BL * KP * 2);
  unsigned short* wqT  = (unsigned short*)take((size_t)H_ * KP * KP * 2);
  unsigned short* wkT  = (unsigned short*)take((size_t)H_ * KP * KP * 2);
  unsigned short* wvT  = (unsigned short*)take((size_t)H_ * KP * KP * 2);
  unsigned short* wfT  = (unsigned short*)take((size_t)KP * HD * 2);
  unsigned short* w1T  = (unsigned short*)take((size_t)KP * KP * 2);
  unsigned short* w2T  = (unsigned short*)take((size_t)KP * KP * 2);
  unsigned short* qbf  = (unsigned short*)take((size_t)BH * L_ * KP * 2);
  unsigned short* kbf  = (unsigned short*)take((size_t)BH * L_ * KP * 2);
  unsigned short* vT   = (unsigned short*)take((size_t)BH * KP * L_ * 2);
  float*          Sbuf = (float*)take((size_t)BH * L_ * L_ * 4);
  unsigned short* Pbf  = (unsigned short*)take((size_t)BH * L_ * L_ * 2);
  unsigned short* cbf  = (unsigned short*)take((size_t)BL * HD * 2);
  float*          zpre = (float*)take((size_t)BL * KP * 4);
  float*          zbuf = (float*)take((size_t)BL * KP * 4);
  unsigned short* zbf  = (unsigned short*)take((size_t)BL * KP * 2);
  unsigned short* f1bf = (unsigned short*)take((size_t)BL * KP * 2);
  float*          f2   = (float*)take((size_t)BL * KP * 4);

  // Only layer ST-1 contributes to the output (h never updated, out overwritten).
  const int LAYER = 5;
  const float* Wq5 = Wq + (size_t)LAYER * H_ * E_ * D_;
  const float* Wk5 = Wk + (size_t)LAYER * H_ * E_ * D_;
  const float* Wv5 = Wv + (size_t)LAYER * H_ * E_ * D_;
  const float* Wf5 = Wf + (size_t)LAYER * HD * D_;
  const float* W15 = W1 + (size_t)LAYER * D_ * D_;
  const float* W25 = W2 + (size_t)LAYER * D_ * D_;
  const float* bq5 = bq + (size_t)LAYER * H_ * D_;
  const float* bk5 = bk + (size_t)LAYER * H_ * D_;
  const float* bv5 = bv + (size_t)LAYER * H_ * D_;
  const float* bf5 = bff + (size_t)LAYER * D_;
  const float* b15 = b1 + (size_t)LAYER * D_;
  const float* b25 = b2 + (size_t)LAYER * D_;

  // 1) embedding + pos encoding -> h (f32), hbf (bf16, K-padded)
  embed_k<<<B_, 256, 0, stream>>>(y, emb, h, hbf);

  // 2) weight conversion (transpose + pad + bf16)
  conv_qkv_k<<<(H_ * KP * KP + 255) / 256, 256, 0, stream>>>(Wq5, wqT);
  conv_qkv_k<<<(H_ * KP * KP + 255) / 256, 256, 0, stream>>>(Wk5, wkT);
  conv_qkv_k<<<(H_ * KP * KP + 255) / 256, 256, 0, stream>>>(Wv5, wvT);
  conv_wf_k<<<(KP * HD + 255) / 256, 256, 0, stream>>>(Wf5, wfT);
  conv_sq_k<<<(KP * KP + 255) / 256, 256, 0, stream>>>(W15, w1T);
  conv_sq_k<<<(KP * KP + 255) / 256, 256, 0, stream>>>(W25, w2T);

  auto launch = [&](auto kern, GemmArgs& P, int Z) {
    int Mt = P.M >> 5;
    int tiles = Mt * P.Ntiles;
    dim3 g((tiles + 7) / 8, 1, Z);
    kern<<<g, 256, 0, stream>>>(P);
  };

  // 3) Q = h*WqT + bq  -> bf16 [b,h,l,224]
  {
    GemmArgs P{};
    P.A = hbf; P.lda = KP; P.zsA = 0;
    P.BT = wqT; P.ldb = KP; P.zsB = (long)KP * KP;
    P.M = BL; P.Ntiles = KP / 32; P.Kpad = KP; P.Nstore = KP;
    P.bias = bq5; P.biasZs = D_; P.biasN = D_;
    P.outB = qbf; P.ldo = KP; P.H = H_; P.L = L_;
    launch(gemm_wmma_k<0>, P, H_);
    P.BT = wkT; P.bias = bk5; P.outB = kbf;
    launch(gemm_wmma_k<0>, P, H_);
    // V -> transposed bf16 [b,h,d(224),l]; rows d>=200 are exact zeros
    P.BT = wvT; P.bias = bv5; P.outB = vT; P.ldo = L_; P.Nstore = KP;
    launch(gemm_wmma_k<1>, P, H_);
  }

  // 4) S = Q*K^T (f32), per (b,h)
  {
    GemmArgs P{};
    P.A = qbf; P.lda = KP; P.zsA = (long)L_ * KP;
    P.BT = kbf; P.ldb = KP; P.zsB = (long)L_ * KP;
    P.M = L_; P.Ntiles = L_ / 32; P.Kpad = KP; P.Nstore = L_;
    P.outF = Sbuf; P.ldo = L_; P.zsO = (long)L_ * L_;
    P.H = H_; P.L = L_;
    launch(gemm_wmma_k<2>, P, BH);
  }

  // 5) P = softmax(S / sqrt(D)) -> bf16
  { dim3 g(L_, BH); softmax_k<<<g, 256, 0, stream>>>(Sbuf, Pbf); }

  // 6) C = P*V -> bf16 [b*l, h*200+d]
  {
    GemmArgs P{};
    P.A = Pbf; P.lda = L_; P.zsA = (long)L_ * L_;
    P.BT = vT; P.ldb = L_; P.zsB = (long)KP * L_;
    P.M = L_; P.Ntiles = KP / 32; P.Kpad = L_; P.Nstore = D_;
    P.outB = cbf; P.ldo = HD; P.H = H_; P.L = L_;
    launch(gemm_wmma_k<3>, P, BH);
  }

  // 7) zpre = C*WfT + bf + h (f32)
  {
    GemmArgs P{};
    P.A = cbf; P.lda = HD; P.BT = wfT; P.ldb = HD;
    P.M = BL; P.Ntiles = KP / 32; P.Kpad = HD; P.Nstore = D_;
    P.bias = bf5; P.biasZs = 0; P.biasN = D_;
    P.outF = zpre; P.ldo = KP; P.res = h; P.ldr = E_;
    P.H = H_; P.L = L_;
    launch(gemm_wmma_k<4>, P, 1);
  }

  // 8) z = LN(zpre) -> f32 [16384,224] + bf16 [16384,224]
  ln_k<<<B_, 256, 0, stream>>>(zpre, KP, zbuf, KP, zbf, KP, L_, D_, KP);

  // 9) f1 = relu(z*W1T + b1) -> bf16 (full 224 cols; pads compute to 0)
  {
    GemmArgs P{};
    P.A = zbf; P.lda = KP; P.BT = w1T; P.ldb = KP;
    P.M = BL; P.Ntiles = KP / 32; P.Kpad = KP; P.Nstore = KP;
    P.bias = b15; P.biasZs = 0; P.biasN = D_;
    P.outB = f1bf; P.ldo = KP; P.H = H_; P.L = L_;
    launch(gemm_wmma_k<5>, P, 1);
  }

  // 10) f2 = f1*W2T + b2 + z (f32)
  {
    GemmArgs P{};
    P.A = f1bf; P.lda = KP; P.BT = w2T; P.ldb = KP;
    P.M = BL; P.Ntiles = KP / 32; P.Kpad = KP; P.Nstore = D_;
    P.bias = b25; P.biasZs = 0; P.biasN = D_;
    P.outF = f2; P.ldo = KP; P.res = zbuf; P.ldr = KP;
    P.H = H_; P.L = L_;
    launch(gemm_wmma_k<6>, P, 1);
  }

  // 11) out = LN(f2) -> d_out fp32 [32,512,200]
  ln_k<<<B_, 256, 0, stream>>>(f2, KP, (float*)d_out, D_, nullptr, 0, L_, D_, 0);
}
